// AmplifiedAttention_60610578481403
// MI455X (gfx1250) — compile-verified, matrix operand order checked by reading, not run on hardware
//
#include <hip/hip_runtime.h>
#include <hip/hip_bf16.h>
#include <math.h>

// ---------------------------------------------------------------------------
// AmplifiedAttention for MI455X (gfx1250, wave32, WMMA bf16 16x16x32).
// out2 == out1 identity exploited (rotate_half cancels in both score terms).
// Attention K/K2/V tiles staged via CDNA5 async global->LDS (ASYNCcnt).
// ---------------------------------------------------------------------------

typedef __bf16 bf16;
typedef __attribute__((ext_vector_type(16))) __bf16 v16bf;
typedef __attribute__((ext_vector_type(8)))  __bf16 v8bf;
typedef __attribute__((ext_vector_type(8)))  float  v8f;

#define S_LEN  2048
#define DMODEL 1024
#define NHEADS 16
#define HDIM   64

// Load one bf16 A/B fragment (16 halves/lane) for 16x16x32 WMMA.
// Per ISA 7.12.2: lane = (r, g), halves e=0..7 -> K = k0 + g*8 + e,
// halves e=8..15 -> K = k0 + 16 + g*8 + (e-8).  Caller passes p already
// offset by row*ld + k0 + g*8; we read 16B at p and 16B at p+16 elems.
__device__ __forceinline__ v16bf load_frag(const bf16* p) {
  v8bf lo = *reinterpret_cast<const v8bf*>(p);
  v8bf hi = *reinterpret_cast<const v8bf*>(p + 16);
  v16bf a;
#pragma unroll
  for (int i = 0; i < 8; ++i) { a[i] = lo[i]; a[i + 8] = hi[i]; }
  return a;
}

__device__ __forceinline__ v8f wmma_bf(v16bf a, v16bf b, v8f c) {
  // D = A(16x32) * B(32x16) + C, f32 accumulate.
  return __builtin_amdgcn_wmma_f32_16x16x32_bf16(
      /*neg_a=*/false, a, /*neg_b=*/false, b,
      /*c_mod=*/(short)0, c, /*reuse_a=*/false, /*reuse_b=*/false);
}

// CDNA5 async global->LDS copy, 16B per lane (ISA 08_async_tensor GV mode):
// VDST = VGPR with LDS byte offset, VADDR = 64-bit global address.
__device__ __forceinline__ void async_g2l_b128(unsigned lds_off,
                                               const void* gaddr) {
  asm volatile("global_load_async_to_lds_b128 %0, %1, off"
               :: "v"(lds_off), "v"(gaddr) : "memory");
}
__device__ __forceinline__ void wait_async0() {
  asm volatile("s_wait_asynccnt 0" ::: "memory");
}

// ---------------------------------------------------------------------------
// Kernel 0: f32 -> bf16 elementwise convert.
// ---------------------------------------------------------------------------
__global__ void __launch_bounds__(256)
cvt_bf16_kernel(const float* __restrict__ in, bf16* __restrict__ out, int n) {
  int i = blockIdx.x * blockDim.x + threadIdx.x;
  if (i < n) out[i] = (bf16)in[i];
}

// ---------------------------------------------------------------------------
// Kernel 1: fused QKV projection (Y = x @ W^T) + RoPE + squares + V-transpose.
// Wave computes a 16(M) x 64(N) strip covering one full head of one matrix.
// grid = (64, 48): x -> 256 row tiles (4 waves/block), y -> mat(3) * head(16).
// ---------------------------------------------------------------------------
__global__ void __launch_bounds__(128)
qkv_rope_kernel(const bf16* __restrict__ xb,
                const bf16* __restrict__ wq, const bf16* __restrict__ wk,
                const bf16* __restrict__ wv,
                bf16* __restrict__ Qr, bf16* __restrict__ Kr,
                bf16* __restrict__ Q2, bf16* __restrict__ K2,
                bf16* __restrict__ Vt) {
  const int wave = threadIdx.x >> 5;
  const int lane = threadIdx.x & 31;
  const int r = lane & 15;
  const int g = lane >> 4;
  const int m0  = (blockIdx.x * 4 + wave) * 16;
  const int mat = blockIdx.y >> 4;          // 0=Q 1=K 2=V
  const int h   = blockIdx.y & 15;
  const bf16* W = (mat == 0) ? wq : (mat == 1) ? wk : wv;
  const int n0 = h * HDIM;

  const bf16* ap = xb + (size_t)(m0 + r) * DMODEL + g * 8;
  const bf16* bp[4];
#pragma unroll
  for (int t = 0; t < 4; ++t)
    bp[t] = W + (size_t)(n0 + t * 16 + r) * DMODEL + g * 8;

  v8f acc[4];
#pragma unroll
  for (int t = 0; t < 4; ++t)
#pragma unroll
    for (int v = 0; v < 8; ++v) acc[t][v] = 0.f;

#pragma unroll 4
  for (int k0 = 0; k0 < DMODEL; k0 += 32) {
    __builtin_prefetch(ap + k0 + 128, 0, 3);
    v16bf a = load_frag(ap + k0);
#pragma unroll
    for (int t = 0; t < 4; ++t)
      acc[t] = wmma_bf(a, load_frag(bp[t] + k0), acc[t]);
  }

  if (mat < 2) {
    bf16* dst  = (mat == 0) ? Qr : Kr;
    bf16* dst2 = (mat == 0) ? Q2 : K2;
#pragma unroll
    for (int t = 0; t < 2; ++t) {
      const int j = t * 16 + r;                           // 0..31 (first half)
      // inv_freq = 10000^(-j/32) = exp(-j * ln(10000)/32)
      const float freq = __expf(-(float)j * (9.210340371976184f / 32.0f));
#pragma unroll
      for (int v = 0; v < 8; ++v) {
        const int row = m0 + v + 8 * g;                   // C-tile: M = v + 8g
        const int b = row >> 11;
        const int s = row & (S_LEN - 1);
        float sn, cs;
        __sincosf((float)s * freq, &sn, &cs);
        const float c1 = acc[t][v];                       // dim j
        const float c2 = acc[t + 2][v];                   // dim j+32
        const float o1 = c1 * cs - c2 * sn;
        const float o2 = c2 * cs + c1 * sn;
        const size_t base = ((size_t)(b * NHEADS + h) * S_LEN + s) * HDIM;
        dst [base + j]      = (bf16)o1;
        dst [base + j + 32] = (bf16)o2;
        dst2[base + j]      = (bf16)(o1 * o1);
        dst2[base + j + 32] = (bf16)(o2 * o2);
      }
    }
  } else {  // V: store transposed [bh][dim][s] so PV B-frags are contiguous
#pragma unroll
    for (int t = 0; t < 4; ++t) {
      const int j = t * 16 + r;
#pragma unroll
      for (int v = 0; v < 8; ++v) {
        const int row = m0 + v + 8 * g;
        const int b = row >> 11;
        const int s = row & (S_LEN - 1);
        Vt[((size_t)(b * NHEADS + h) * HDIM + j) * S_LEN + s] = (bf16)acc[t][v];
      }
    }
  }
}

// ---------------------------------------------------------------------------
// Kernel 2: causal flash attention with amplified score + fused gate epilogue.
// grid = (32 qblocks, 32 bh); block = 4 waves; wave owns 16 query rows.
// K/K2/V tiles (64x64 bf16 each) staged once per block via async->LDS;
// score = 0.125*QK^T + 0.0125*(Q^2)(K^2)^T ; o_final = o + 0.05 o^2 gate_w.
// ---------------------------------------------------------------------------
__global__ void __launch_bounds__(128)
attn_kernel(const bf16* __restrict__ Qr, const bf16* __restrict__ Kr,
            const bf16* __restrict__ Q2, const bf16* __restrict__ K2,
            const bf16* __restrict__ Vt, const float* __restrict__ gate_w,
            bf16* __restrict__ G) {
  __shared__ __align__(16) bf16 Kbuf[64][HDIM];      // 8 KB  [key][dim]
  __shared__ __align__(16) bf16 K2buf[64][HDIM];     // 8 KB
  __shared__ __align__(16) bf16 Vbuf[64][64];        // 8 KB  [dim][key]
  __shared__ __align__(16) bf16 Pbuf[4][16][HDIM];   // 8 KB  per-wave P

  const int tid  = threadIdx.x;
  const int wave = tid >> 5;
  const int lane = tid & 31;
  const int r = lane & 15;
  const int g = lane >> 4;
  const int bh = blockIdx.y;
  const int qb = blockIdx.x;
  const size_t hoff = (size_t)bh * S_LEN * HDIM;
  const bf16* Qp  = Qr + hoff;
  const bf16* Kp  = Kr + hoff;
  const bf16* Q2p = Q2 + hoff;
  const bf16* K2p = K2 + hoff;
  const bf16* Vp  = Vt + hoff;                       // [dim][s]
  const int m0 = qb * 64 + wave * 16;

  const unsigned kb_base  = (unsigned)(uintptr_t)&Kbuf[0][0];
  const unsigned k2b_base = (unsigned)(uintptr_t)&K2buf[0][0];
  const unsigned vb_base  = (unsigned)(uintptr_t)&Vbuf[0][0];

  // Q fragments live in registers for the whole pass (hd=64 -> 2 K-steps).
  v16bf aq[2], aq2[2];
#pragma unroll
  for (int ks = 0; ks < 2; ++ks) {
    aq[ks]  = load_frag(Qp  + (size_t)(m0 + r) * HDIM + ks * 32 + g * 8);
    aq2[ks] = load_frag(Q2p + (size_t)(m0 + r) * HDIM + ks * 32 + g * 8);
  }

  v8f o[4];
#pragma unroll
  for (int t = 0; t < 4; ++t)
#pragma unroll
    for (int v = 0; v < 8; ++v) o[t][v] = 0.f;
  float mrow[8], lrow[8], gw[4];
#pragma unroll
  for (int v = 0; v < 8; ++v) { mrow[v] = -3.0e38f; lrow[v] = 0.f; }
#pragma unroll
  for (int t = 0; t < 4; ++t) gw[t] = gate_w[t * 16 + r];

  for (int kb = 0; kb <= qb; ++kb) {                 // causal: skip upper blocks
    const int t0 = kb * 64;

    // ---- cooperative async stage of K / K2 / V 64x64 tiles into LDS ----
    // 512 x 16B chunks per tile; 128 threads x 4 chunks each.
#pragma unroll
    for (int i = 0; i < 4; ++i) {
      const int c   = tid + i * 128;                 // 0..511
      const int row = c >> 3;                        // tile row
      const int col = c & 7;                         // 16B chunk in row
      const size_t gofs = (size_t)(t0 + row) * HDIM + col * 8;
      async_g2l_b128(kb_base  + c * 16, Kp  + gofs);
      async_g2l_b128(k2b_base + c * 16, K2p + gofs);
      const size_t vofs = (size_t)row * S_LEN + t0 + col * 8;  // row = dim
      async_g2l_b128(vb_base  + c * 16, Vp + vofs);
    }
    wait_async0();
    __syncthreads();                                 // B1: tiles visible

    v8f s1[4], s2[4];
#pragma unroll
    for (int t = 0; t < 4; ++t)
#pragma unroll
      for (int v = 0; v < 8; ++v) { s1[t][v] = 0.f; s2[t][v] = 0.f; }

#pragma unroll
    for (int ks = 0; ks < 2; ++ks) {
#pragma unroll
      for (int tt = 0; tt < 4; ++tt) {
        const int krow = tt * 16 + r;
        s1[tt] = wmma_bf(aq[ks],
                         load_frag(&Kbuf[krow][ks * 32 + g * 8]), s1[tt]);
        s2[tt] = wmma_bf(aq2[ks],
                         load_frag(&K2buf[krow][ks * 32 + g * 8]), s2[tt]);
      }
    }

    float pv[4][8], lm[8];
#pragma unroll
    for (int v = 0; v < 8; ++v) lm[v] = -3.0e38f;
#pragma unroll
    for (int tt = 0; tt < 4; ++tt) {
#pragma unroll
      for (int v = 0; v < 8; ++v) {
        float sc = 0.125f * s1[tt][v] + 0.0125f * s2[tt][v];
        const int tcol = t0 + tt * 16 + r;
        const int srow = m0 + v + 8 * g;
        if (tcol > srow) sc = -1.0e9f;               // causal mask
        pv[tt][v] = sc;
        lm[v] = fmaxf(lm[v], sc);
      }
    }
    // row reductions across the 16 lanes holding each row
#pragma unroll
    for (int v = 0; v < 8; ++v)
#pragma unroll
      for (int off = 1; off < 16; off <<= 1)
        lm[v] = fmaxf(lm[v], __shfl_xor(lm[v], off, 32));

#pragma unroll
    for (int v = 0; v < 8; ++v) {
      const float mnew  = fmaxf(mrow[v], lm[v]);
      const float alpha = __expf(mrow[v] - mnew);
      mrow[v] = mnew;
      float rs = 0.f;
#pragma unroll
      for (int tt = 0; tt < 4; ++tt) {
        const float p = __expf(pv[tt][v] - mnew);
        pv[tt][v] = p;
        rs += p;
      }
#pragma unroll
      for (int off = 1; off < 16; off <<= 1)
        rs += __shfl_xor(rs, off, 32);
      lrow[v] = lrow[v] * alpha + rs;
#pragma unroll
      for (int td = 0; td < 4; ++td) o[td][v] *= alpha;
    }

    // Re-layout P: C-tile layout -> A-fragment layout, via LDS.
#pragma unroll
    for (int tt = 0; tt < 4; ++tt)
#pragma unroll
      for (int v = 0; v < 8; ++v)
        Pbuf[wave][v + 8 * g][tt * 16 + r] = (bf16)pv[tt][v];
    __syncthreads();                                 // B2: P visible

#pragma unroll
    for (int ks = 0; ks < 2; ++ks) {
      v16bf ap = load_frag(&Pbuf[wave][r][ks * 32 + g * 8]);
#pragma unroll
      for (int td = 0; td < 4; ++td) {
        const v16bf bv = load_frag(&Vbuf[td * 16 + r][ks * 32 + g * 8]);
        o[td] = wmma_bf(ap, bv, o[td]);
      }
    }
    __syncthreads();                                 // B3: reads done before
  }                                                  //     next async overwrite

  // Epilogue: normalize, gate (out1==out2 => o + 0.05*o^2*g), store bf16.
  const int b = bh >> 4, h = bh & 15;
#pragma unroll
  for (int td = 0; td < 4; ++td) {
#pragma unroll
    for (int v = 0; v < 8; ++v) {
      const float oo = o[td][v] / lrow[v];
      const float of = oo + 0.05f * oo * oo * gw[td];
      const int srow = m0 + v + 8 * g;
      G[(size_t)(b * S_LEN + srow) * DMODEL + h * HDIM + td * 16 + r] = (bf16)of;
    }
  }
}

// ---------------------------------------------------------------------------
// Kernel 3: final projection Out = G @ Wo^T (fp32 out). grid = (64, 16).
// ---------------------------------------------------------------------------
__global__ void __launch_bounds__(128)
out_proj_kernel(const bf16* __restrict__ Gm, const bf16* __restrict__ wo,
                float* __restrict__ out) {
  const int wave = threadIdx.x >> 5;
  const int lane = threadIdx.x & 31;
  const int r = lane & 15;
  const int g = lane >> 4;
  const int m0 = (blockIdx.x * 4 + wave) * 16;
  const int n0 = blockIdx.y * 64;

  const bf16* ap = Gm + (size_t)(m0 + r) * DMODEL + g * 8;
  const bf16* bp[4];
#pragma unroll
  for (int t = 0; t < 4; ++t)
    bp[t] = wo + (size_t)(n0 + t * 16 + r) * DMODEL + g * 8;

  v8f acc[4];
#pragma unroll
  for (int t = 0; t < 4; ++t)
#pragma unroll
    for (int v = 0; v < 8; ++v) acc[t][v] = 0.f;

#pragma unroll 4
  for (int k0 = 0; k0 < DMODEL; k0 += 32) {
    __builtin_prefetch(ap + k0 + 128, 0, 3);
    v16bf a = load_frag(ap + k0);
#pragma unroll
    for (int t = 0; t < 4; ++t)
      acc[t] = wmma_bf(a, load_frag(bp[t] + k0), acc[t]);
  }

#pragma unroll
  for (int t = 0; t < 4; ++t)
#pragma unroll
    for (int v = 0; v < 8; ++v)
      out[(size_t)(m0 + v + 8 * g) * DMODEL + n0 + t * 16 + r] = acc[t][v];
}

// ---------------------------------------------------------------------------
// Host-side launcher.
// ---------------------------------------------------------------------------
extern "C" void kernel_launch(void* const* d_in, const int* in_sizes, int n_in,
                              void* d_out, int out_size, void* d_ws,
                              size_t ws_size, hipStream_t stream) {
  (void)in_sizes; (void)n_in; (void)out_size; (void)ws_size;
  const float* x  = (const float*)d_in[0];
  const float* Wq = (const float*)d_in[1];
  const float* Wk = (const float*)d_in[2];
  const float* Wv = (const float*)d_in[3];
  const float* Wo = (const float*)d_in[4];
  const float* gw = (const float*)d_in[5];

  char* ws = (char*)d_ws;
  const size_t MB = 1024ull * 1024ull;
  bf16* xb  = (bf16*)(ws +  0 * MB);   // 8 MB  [4096][1024] bf16
  bf16* wqb = (bf16*)(ws +  8 * MB);   // 2 MB
  bf16* wkb = (bf16*)(ws + 10 * MB);   // 2 MB
  bf16* wvb = (bf16*)(ws + 12 * MB);   // 2 MB
  bf16* wob = (bf16*)(ws + 14 * MB);   // 2 MB
  bf16* Qr  = (bf16*)(ws + 16 * MB);   // 8 MB  [32][2048][64]
  bf16* Kr  = (bf16*)(ws + 24 * MB);   // 8 MB
  bf16* Q2  = (bf16*)(ws + 32 * MB);   // 8 MB
  bf16* K2  = (bf16*)(ws + 40 * MB);   // 8 MB
  bf16* Vt  = (bf16*)(ws + 48 * MB);   // 8 MB  [32][64][2048]
  bf16* G   = (bf16*)(ws + 56 * MB);   // 8 MB  [4096][1024]

  const int NX = 2 * S_LEN * DMODEL;   // 4194304
  const int NW = DMODEL * DMODEL;      // 1048576
  cvt_bf16_kernel<<<(NX + 255) / 256, 256, 0, stream>>>(x,  xb,  NX);
  cvt_bf16_kernel<<<(NW + 255) / 256, 256, 0, stream>>>(Wq, wqb, NW);
  cvt_bf16_kernel<<<(NW + 255) / 256, 256, 0, stream>>>(Wk, wkb, NW);
  cvt_bf16_kernel<<<(NW + 255) / 256, 256, 0, stream>>>(Wv, wvb, NW);
  cvt_bf16_kernel<<<(NW + 255) / 256, 256, 0, stream>>>(Wo, wob, NW);

  qkv_rope_kernel<<<dim3(64, 48), 128, 0, stream>>>(xb, wqb, wkb, wvb,
                                                    Qr, Kr, Q2, K2, Vt);
  attn_kernel<<<dim3(32, 32), 128, 0, stream>>>(Qr, Kr, Q2, K2, Vt, gw, G);
  out_proj_kernel<<<dim3(64, 16), 128, 0, stream>>>(G, wob, (float*)d_out);
}
